// SinkhornDivergenceFunctional_40587440947655
// MI455X (gfx1250) — matrix operand order repaired and checked by reading, not compiled, not run on hardware
//
#include <hip/hip_runtime.h>
#include <cmath>

#define D_DIM 64
#define WAVES 8

typedef __bf16 bf16;
typedef __attribute__((ext_vector_type(16))) __bf16 v16bf;
typedef __attribute__((ext_vector_type(8)))  __bf16 v8bf;
typedef __attribute__((ext_vector_type(8)))  float  v8f;

union V16U { v16bf v; v8bf h[2]; };

// ---------------------------------------------------------------------------
// Phase 0: per-row half-squared-norms + fp32 -> bf16 hi/lo split (one wave/row)
// ---------------------------------------------------------------------------
__global__ __launch_bounds__(256) void sk_prep_split(
    const float* __restrict__ X, unsigned short* __restrict__ hi,
    unsigned short* __restrict__ lo, float* __restrict__ halfnorm, int nrows)
{
  const int gw   = (int)((blockIdx.x * blockDim.x + threadIdx.x) >> 5);
  const int lane = threadIdx.x & 31;
  if (gw >= nrows) return;
  const float* row = X + (size_t)gw * D_DIM;
  float ssq = 0.0f;
#pragma unroll
  for (int t = 0; t < 2; ++t) {
    const int k   = lane + 32 * t;
    const float v = row[k];
    ssq = fmaf(v, v, ssq);
    const bf16 h = (bf16)v;                 // RNE to bf16
    const bf16 l = (bf16)(v - (float)h);    // residual, next ~8 mantissa bits
    hi[(size_t)gw * D_DIM + k] = __builtin_bit_cast(unsigned short, h);
    lo[(size_t)gw * D_DIM + k] = __builtin_bit_cast(unsigned short, l);
  }
#pragma unroll
  for (int off = 16; off > 0; off >>= 1)
    ssq += __shfl_xor(ssq, off, 32);
  if (lane == 0) halfnorm[gw] = 0.5f * ssq;
}

// ---------------------------------------------------------------------------
// Fused cost + softmin:
//   out[i] = -eps * logsumexp_j( base_log + use_pot*pot[j]/eps - C_ij/eps )
//   C_ij   = max(sa[i] + sb[j] - A_i . B_j, 0)
// A_i.B_j via 8x bf16 WMMA (hi/lo split, two independent 4-deep chains).
// Block = 8 waves = 16 output rows; wave w streams j-tiles w, w+8, ...
// ---------------------------------------------------------------------------
__global__ __launch_bounds__(256) void sk_softmin_wmma(
    const unsigned short* __restrict__ Ahi, const unsigned short* __restrict__ Alo,
    const float* __restrict__ sa,
    const unsigned short* __restrict__ Bhi, const unsigned short* __restrict__ Blo,
    const float* __restrict__ sb,
    const float* __restrict__ pot, float use_pot, float base_log,
    const float* __restrict__ eps_list, int eps_idx,
    float* __restrict__ out, int Mcols)
{
  __shared__ float lmax[WAVES][16];
  __shared__ float lsum[WAVES][16];

  const int lane = threadIdx.x & 31;
  const int wave = threadIdx.x >> 5;
  const int n    = lane & 15;
  const int half = lane >> 4;
  const int i0   = blockIdx.x << 4;

  const float eps       = eps_list[eps_idx];     // uniform -> s_load
  const float inv_eps   = 1.0f / eps;
  const float pot_scale = use_pot * inv_eps;     // 0 disables the potential

  // ---- A operand (16 rows x 64 K), VGPR-resident for the whole j sweep ----
  // A k order used here (contiguous, same as B): lane(m=lane&15, h=half) holds K = kb*32 + 16h + 0..15
  V16U a_hi[2], a_lo[2];
  {
    const size_t rbase = (size_t)(i0 + n) * D_DIM;
#pragma unroll
    for (int kb = 0; kb < 2; ++kb) {
      const int kofs = 32 * kb + 16 * half;
      a_hi[kb].h[0] = *(const v8bf*)(const void*)(Ahi + rbase + kofs);
      a_hi[kb].h[1] = *(const v8bf*)(const void*)(Ahi + rbase + kofs + 8);
      a_lo[kb].h[0] = *(const v8bf*)(const void*)(Alo + rbase + kofs);
      a_lo[kb].h[1] = *(const v8bf*)(const void*)(Alo + rbase + kofs + 8);
    }
  }

  // C/D layout: lane(n, half) holds rows 8*half + r, r = 0..7
  float sar[8];
#pragma unroll
  for (int r = 0; r < 8; ++r) sar[r] = sa[i0 + 8 * half + r];

  float mx[8], sm[8];
#pragma unroll
  for (int r = 0; r < 8; ++r) { mx[r] = -__builtin_inff(); sm[r] = 0.0f; }

  const int ntiles = Mcols >> 4;
  for (int jt = wave; jt < ntiles; jt += WAVES) {
    const int j = (jt << 4) + n;                    // this lane's column
    // B k order used here (contiguous, same as A): lane(n, half) holds K = kb*32 + 16*half + 0..15
    // => one contiguous 32B run per kb: row n of Y, plain b128 loads.
    const size_t cbase = (size_t)j * D_DIM + 16 * half;
    V16U b_hi[2], b_lo[2];
    b_hi[0].v = *(const v16bf*)(const void*)(Bhi + cbase);
    b_hi[1].v = *(const v16bf*)(const void*)(Bhi + cbase + 32);
    b_lo[0].v = *(const v16bf*)(const void*)(Blo + cbase);
    b_lo[1].v = *(const v16bf*)(const void*)(Blo + cbase + 32);

    // issue the scalar epilogue loads early so they overlap the WMMAs
    const float sbj  = sb[j];
    const float potj = pot[j];

    // speculative prefetch of the next tile this wave owns (safe past the end:
    // SE-scope prefetch translation failures are silently dropped)
    __builtin_prefetch(Bhi + cbase + (size_t)16 * WAVES * D_DIM, 0, 3);
    __builtin_prefetch(Blo + cbase + (size_t)16 * WAVES * D_DIM, 0, 3);

    // two independent accumulation chains (kb=0 / kb=1) for WMMA ILP
    v8f acc0 = {0.f, 0.f, 0.f, 0.f, 0.f, 0.f, 0.f, 0.f};
    v8f acc1 = {0.f, 0.f, 0.f, 0.f, 0.f, 0.f, 0.f, 0.f};
    acc0 = __builtin_amdgcn_wmma_f32_16x16x32_bf16(false, a_lo[0].v, false, b_lo[0].v, (short)0, acc0, false, false);
    acc1 = __builtin_amdgcn_wmma_f32_16x16x32_bf16(false, a_lo[1].v, false, b_lo[1].v, (short)0, acc1, false, false);
    acc0 = __builtin_amdgcn_wmma_f32_16x16x32_bf16(false, a_lo[0].v, false, b_hi[0].v, (short)0, acc0, false, false);
    acc1 = __builtin_amdgcn_wmma_f32_16x16x32_bf16(false, a_lo[1].v, false, b_hi[1].v, (short)0, acc1, false, false);
    acc0 = __builtin_amdgcn_wmma_f32_16x16x32_bf16(false, a_hi[0].v, false, b_lo[0].v, (short)0, acc0, false, false);
    acc1 = __builtin_amdgcn_wmma_f32_16x16x32_bf16(false, a_hi[1].v, false, b_lo[1].v, (short)0, acc1, false, false);
    acc0 = __builtin_amdgcn_wmma_f32_16x16x32_bf16(false, a_hi[0].v, false, b_hi[0].v, (short)0, acc0, false, false);
    acc1 = __builtin_amdgcn_wmma_f32_16x16x32_bf16(false, a_hi[1].v, false, b_hi[1].v, (short)0, acc1, false, false);

    const float hj = fmaf(potj, pot_scale, base_log);  // branchless
#pragma unroll
    for (int r = 0; r < 8; ++r) {
      float c = sar[r] + sbj - (acc0[r] + acc1[r]);
      c = fmaxf(c, 0.0f);
      const float t  = fmaf(-c, inv_eps, hj);
      const float mo = mx[r];
      mx[r] = fmaxf(mo, t);
      const float e = __expf(-fabsf(mo - t));          // single exp/element
      sm[r] = (mo >= t) ? (sm[r] + e) : fmaf(sm[r], e, 1.0f);
    }
  }

  // combine the 16 columns held across the 16 lanes of this half-wave
#pragma unroll
  for (int off = 1; off < 16; off <<= 1) {
#pragma unroll
    for (int r = 0; r < 8; ++r) {
      const float mo = __shfl_xor(mx[r], off, 32);
      const float so = __shfl_xor(sm[r], off, 32);
      const float mn = fmaxf(mx[r], mo);
      sm[r] = sm[r] * __expf(mx[r] - mn) + so * __expf(mo - mn);
      mx[r] = mn;
    }
  }

  if (n == 0) {
#pragma unroll
    for (int r = 0; r < 8; ++r) {
      lmax[wave][8 * half + r] = mx[r];
      lsum[wave][8 * half + r] = sm[r];
    }
  }
  __syncthreads();

  if (threadIdx.x < 16) {                           // combine the 8 wave partials
    const int row = threadIdx.x;
    float Mv = lmax[0][row], Sv = lsum[0][row];
#pragma unroll
    for (int w = 1; w < WAVES; ++w) {
      const float mo = lmax[w][row], so = lsum[w][row];
      const float mn = fmaxf(Mv, mo);
      Sv = Sv * __expf(Mv - mn) + so * __expf(mo - mn);
      Mv = mn;
    }
    out[i0 + row] = -eps * (Mv + __logf(Sv));
  }
}

// symmetrized dual update: a = 0.5*(a+at), b = 0.5*(b+bt)
__global__ void sk_avg2(float* __restrict__ a, const float* __restrict__ at,
                        float* __restrict__ b, const float* __restrict__ bt, int nelem)
{
  const int i = blockIdx.x * blockDim.x + threadIdx.x;
  if (i < nelem) {
    a[i] = 0.5f * (a[i] + at[i]);
    b[i] = 0.5f * (b[i] + bt[i]);
  }
}

// result = mean(f_ba_f - f_aa_f) + mean(g_ab_f - g_bb_f)
__global__ __launch_bounds__(256) void sk_finalize(
    const float* __restrict__ fbaf, const float* __restrict__ faaf, int N,
    const float* __restrict__ gabf, const float* __restrict__ gbbf, int M,
    float* __restrict__ out)
{
  __shared__ float sd[256];
  const int tid = threadIdx.x;
  float a = 0.f, b = 0.f;
  for (int i = tid; i < N; i += 256) a += fbaf[i] - faaf[i];
  for (int i = tid; i < M; i += 256) b += gabf[i] - gbbf[i];
  sd[tid] = a / (float)N + b / (float)M;
  __syncthreads();
  for (int s = 128; s > 0; s >>= 1) {
    if (tid < s) sd[tid] += sd[tid + s];
    __syncthreads();
  }
  if (tid == 0) out[0] = sd[0];
}

// ---------------------------------------------------------------------------
static inline void launch_softmin(hipStream_t stream,
    const unsigned short* Ahi, const unsigned short* Alo, const float* sa, int nrows,
    const unsigned short* Bhi, const unsigned short* Blo, const float* sb, int mcols,
    const float* pot, float use_pot, float base_log,
    const float* eps_list, int eps_idx, float* out)
{
  sk_softmin_wmma<<<nrows / 16, 256, 0, stream>>>(
      Ahi, Alo, sa, Bhi, Blo, sb, pot, use_pot, base_log, eps_list, eps_idx, out, mcols);
}

extern "C" void kernel_launch(void* const* d_in, const int* in_sizes, int n_in,
                              void* d_out, int out_size, void* d_ws, size_t ws_size,
                              hipStream_t stream)
{
  (void)n_in; (void)out_size; (void)ws_size;
  const float* X        = (const float*)d_in[0];
  const float* Y        = (const float*)d_in[1];
  const float* eps_list = (const float*)d_in[2];
  const int n_eps = in_sizes[2];
  const int N = in_sizes[0] / D_DIM;
  const int M = in_sizes[1] / D_DIM;

  char* p = (char*)d_ws;
  auto carve = [&](size_t bytes) -> void* {
    void* q = (void*)p; p += (bytes + 255) & ~(size_t)255; return q;
  };
  unsigned short* xhi = (unsigned short*)carve((size_t)N * D_DIM * 2);
  unsigned short* xlo = (unsigned short*)carve((size_t)N * D_DIM * 2);
  unsigned short* yhi = (unsigned short*)carve((size_t)M * D_DIM * 2);
  unsigned short* ylo = (unsigned short*)carve((size_t)M * D_DIM * 2);
  float* sx   = (float*)carve((size_t)N * 4);
  float* sy   = (float*)carve((size_t)M * 4);
  float* f_aa = (float*)carve((size_t)N * 4);
  float* f_ba = (float*)carve((size_t)N * 4);
  float* g_bb = (float*)carve((size_t)M * 4);
  float* g_ab = (float*)carve((size_t)M * 4);
  float* t_ft  = (float*)carve((size_t)N * 4);
  float* t_faa = (float*)carve((size_t)N * 4);
  float* t_gt  = (float*)carve((size_t)M * 4);
  float* t_gbb = (float*)carve((size_t)M * 4);

  const float a_log = -logf((float)N);
  const float b_log = -logf((float)M);

  sk_prep_split<<<(N * 32 + 255) / 256, 256, 0, stream>>>(X, xhi, xlo, sx, N);
  sk_prep_split<<<(M * 32 + 255) / 256, 256, 0, stream>>>(Y, yhi, ylo, sy, M);

  // ---- initialization at eps0 (uniform log-weights; potential disabled) ----
  launch_softmin(stream, xhi, xlo, sx, N, xhi, xlo, sx, N, sx, 0.f, a_log, eps_list, 0, f_aa);
  launch_softmin(stream, yhi, ylo, sy, M, yhi, ylo, sy, M, sy, 0.f, b_log, eps_list, 0, g_bb);
  launch_softmin(stream, yhi, ylo, sy, M, xhi, xlo, sx, N, sx, 0.f, a_log, eps_list, 0, g_ab);
  launch_softmin(stream, xhi, xlo, sx, N, yhi, ylo, sy, M, sy, 0.f, b_log, eps_list, 0, f_ba);

  // ---- eps-scaling Sinkhorn loop (symmetrized updates) ----
  for (int k = 0; k < n_eps; ++k) {
    launch_softmin(stream, xhi, xlo, sx, N, yhi, ylo, sy, M, g_ab, 1.f, b_log, eps_list, k, t_ft);   // ft
    launch_softmin(stream, yhi, ylo, sy, M, xhi, xlo, sx, N, f_ba, 1.f, a_log, eps_list, k, t_gt);   // gt
    launch_softmin(stream, xhi, xlo, sx, N, xhi, xlo, sx, N, f_aa, 1.f, a_log, eps_list, k, t_faa);  // ftaa
    launch_softmin(stream, yhi, ylo, sy, M, yhi, ylo, sy, M, g_bb, 1.f, b_log, eps_list, k, t_gbb);  // gtbb
    sk_avg2<<<(N + 255) / 256, 256, 0, stream>>>(f_aa, t_faa, f_ba, t_ft, N);
    sk_avg2<<<(M + 255) / 256, 256, 0, stream>>>(g_bb, t_gbb, g_ab, t_gt, M);
  }

  // ---- final extrapolation at eps = blur^p ----
  const int kl = n_eps - 1;
  launch_softmin(stream, xhi, xlo, sx, N, yhi, ylo, sy, M, g_ab, 1.f, b_log, eps_list, kl, t_ft);   // f_ba_f
  launch_softmin(stream, yhi, ylo, sy, M, xhi, xlo, sx, N, f_ba, 1.f, a_log, eps_list, kl, t_gt);   // g_ab_f
  launch_softmin(stream, xhi, xlo, sx, N, xhi, xlo, sx, N, f_aa, 1.f, a_log, eps_list, kl, t_faa);  // f_aa_f
  launch_softmin(stream, yhi, ylo, sy, M, yhi, ylo, sy, M, g_bb, 1.f, b_log, eps_list, kl, t_gbb);  // g_bb_f

  sk_finalize<<<1, 256, 0, stream>>>(t_ft, t_faa, N, t_gt, t_gbb, M, (float*)d_out);
}
